// MultiheadedAttention_78512002171369
// MI455X (gfx1250) — matrix operand order repaired and checked by reading, not compiled, not run on hardware
//
#include <hip/hip_runtime.h>
#include <hip/hip_bf16.h>

// MI455X / gfx1250 fused multi-head attention:
//   bf16 pre-convert (X row-major, W transposed) ->
//   proj (bf16 WMMA, f32 acc, 32 rows/wave, scale folded for Q, V transposed) ->
//   flash attention (bf16 WMMA, 32 query rows/wave, online softmax,
//                    all fragments as b128 loads, K/V fragments reused 2x)

#define N_SEQ   4096
#define H_HEADS 16
#define DM      1024
#define DH      64

typedef __attribute__((ext_vector_type(16))) __bf16 v16bf;
typedef __attribute__((ext_vector_type(8)))  float  v8f;

// 16-bit A-fragment (16x32) k-pattern used here (contiguous; ISA 7.12.2 table order differs):
//   elements e=0..7  -> k = hi*16 + e        (VGPR0..3)
//   elements e=8..15 -> k = hi*16 + e (VGPR4..7)
// B-fragment (32x16): lane n = lane&15, element e -> k = hi*16 + e.
// Both are contiguous 8/16-element runs -> 16B loads.

union Frag {                 // 16 bf16 = 32 bytes
    v16bf v;
    uint4 q[2];
};

// ---------------------------------------------------------------------------
// X -> bf16, row-major copy. 8 elements per thread.
// ---------------------------------------------------------------------------
__global__ void __launch_bounds__(256)
cvtx_kernel(const float* __restrict__ X, __bf16* __restrict__ Y) {
    size_t i = (size_t)blockIdx.x * blockDim.x + threadIdx.x;  // < N*DM/8
    const float4* xp = (const float4*)X + 2 * i;
    float4 f0 = xp[0], f1 = xp[1];
    union { __bf16 h8[8]; uint4 q; } u;
    u.h8[0] = (__bf16)f0.x; u.h8[1] = (__bf16)f0.y;
    u.h8[2] = (__bf16)f0.z; u.h8[3] = (__bf16)f0.w;
    u.h8[4] = (__bf16)f1.x; u.h8[5] = (__bf16)f1.y;
    u.h8[6] = (__bf16)f1.z; u.h8[7] = (__bf16)f1.w;
    ((uint4*)Y)[i] = u.q;
}

// ---------------------------------------------------------------------------
// W[h][d][k] -> Wt[h][k][d] bf16 (transpose + convert). 8 d's per thread.
// ---------------------------------------------------------------------------
__global__ void __launch_bounds__(256)
cvtw_kernel(const float* __restrict__ W, __bf16* __restrict__ Wt) {
    int t  = blockIdx.x * blockDim.x + threadIdx.x;  // < H*DH*(DM/8) = 131072
    int d8 = t & (DM / 8 - 1);
    int k  = (t >> 7) & (DH - 1);
    int h  = t >> 13;
    union { __bf16 h8[8]; uint4 q; } u;
    const float* src = W + (size_t)h * DM * DH + (size_t)d8 * 8 * DH + k;
    #pragma unroll
    for (int j = 0; j < 8; ++j) u.h8[j] = (__bf16)src[(size_t)j * DH];
    *(uint4*)&Wt[((size_t)h * DH + k) * DM + d8 * 8] = u.q;
}

// ---------------------------------------------------------------------------
// Projection: acc = Xb . Wt[h]^T (+bias)*scale.
// One wave -> 32 rows x 64 cols for head blockIdx.y (B-fragments reused 2x).
// col_major=0: Out[h][n][k] (for Q,K)   col_major=1: Out[h][k][n] (for V)
// ---------------------------------------------------------------------------
__global__ void __launch_bounds__(128)
proj_kernel(const __bf16* __restrict__ Xb, const __bf16* __restrict__ Wt,
            const float* __restrict__ bias, __bf16* __restrict__ Out,
            float out_scale, int col_major) {
    const int lane = threadIdx.x & 31;
    const int wave = threadIdx.x >> 5;
    const int h    = blockIdx.y;
    const int r0   = (blockIdx.x * 4 + wave) * 32;
    const int m    = lane & 15;
    const int hi   = lane >> 4;

    const __bf16* Wth = Wt + (size_t)h * DH * DM;  // [64][1024]

    v8f acc[2][4];
    #pragma unroll
    for (int mt = 0; mt < 2; ++mt)
        #pragma unroll
        for (int nt = 0; nt < 4; ++nt)
            #pragma unroll
            for (int v = 0; v < 8; ++v) acc[mt][nt][v] = 0.0f;

    const __bf16* xrow0 = Xb + (size_t)(r0 + m) * DM;
    const __bf16* xrow1 = Xb + (size_t)(r0 + 16 + m) * DM;

    for (int d0 = 0; d0 < DM; d0 += 32) {
        Frag a[2];
        a[0].q[0] = *(const uint4*)(xrow0 + d0 + hi * 16);
        a[0].q[1] = *(const uint4*)(xrow0 + d0 + hi * 16 + 8);
        a[1].q[0] = *(const uint4*)(xrow1 + d0 + hi * 16);
        a[1].q[1] = *(const uint4*)(xrow1 + d0 + hi * 16 + 8);
        #pragma unroll
        for (int nt = 0; nt < 4; ++nt) {
            const __bf16* wp = Wth + (size_t)(nt * 16 + m) * DM + d0 + hi * 16;
            Frag b;
            b.q[0] = *(const uint4*)(wp);
            b.q[1] = *(const uint4*)(wp + 8);
            #pragma unroll
            for (int mt = 0; mt < 2; ++mt)
                acc[mt][nt] = __builtin_amdgcn_wmma_f32_16x16x32_bf16(
                    false, a[mt].v, false, b.v, (short)0, acc[mt][nt], false, false);
        }
    }

    #pragma unroll
    for (int mt = 0; mt < 2; ++mt) {
        int rbase = r0 + mt * 16;
        if (col_major) {
            // Out[h][col][row]: 8 consecutive rows per lane -> one b128 store
            #pragma unroll
            for (int nt = 0; nt < 4; ++nt) {
                float bv = bias[h * DH + nt * 16 + m];
                union { __bf16 h8[8]; uint4 q; } st;
                #pragma unroll
                for (int v = 0; v < 8; ++v)
                    st.h8[v] = (__bf16)((acc[mt][nt][v] + bv) * out_scale);
                *(uint4*)&Out[((size_t)h * DH + nt * 16 + m) * N_SEQ +
                              rbase + hi * 8] = st.q;
            }
        } else {
            #pragma unroll
            for (int nt = 0; nt < 4; ++nt) {
                float bv = bias[h * DH + nt * 16 + m];
                #pragma unroll
                for (int v = 0; v < 8; ++v) {
                    int row = rbase + v + hi * 8;
                    Out[((size_t)h * N_SEQ + row) * DH + nt * 16 + m] =
                        (__bf16)((acc[mt][nt][v] + bv) * out_scale);
                }
            }
        }
    }
}

// ---------------------------------------------------------------------------
// Flash attention. One wave per (head, 32-query-row tile), 32-key chunks.
// Qb,Kb row-major [h][n][64] (Q pre-scaled by 1/8); Vt col-major [h][k][n].
// K/V fragments are loaded once per chunk and reused by both row-tiles.
// ---------------------------------------------------------------------------
__global__ void __launch_bounds__(128)
attn_kernel(const __bf16* __restrict__ Qb, const __bf16* __restrict__ Kb,
            const __bf16* __restrict__ Vt, float* __restrict__ Out) {
    __shared__ __bf16 Pld[4][2][16 * 32];   // per-wave, per-row-tile P staging

    const int lane = threadIdx.x & 31;
    const int wave = threadIdx.x >> 5;
    const int h    = blockIdx.y;
    const int q0   = (blockIdx.x * 4 + wave) * 32;
    const int m    = lane & 15;
    const int hi   = lane >> 4;

    const __bf16* Qh  = Qb + (size_t)h * N_SEQ * DH;
    const __bf16* Kh  = Kb + (size_t)h * N_SEQ * DH;
    const __bf16* Vth = Vt + (size_t)h * DH * N_SEQ;

    // Q fragments: [row-tile][k-half]; scale already folded in projection
    Frag aQ[2][2];
    #pragma unroll
    for (int mt = 0; mt < 2; ++mt)
        #pragma unroll
        for (int kh = 0; kh < 2; ++kh) {
            const __bf16* qp = Qh + (size_t)(q0 + mt * 16 + m) * DH + kh * 32;
            aQ[mt][kh].q[0] = *(const uint4*)(qp + hi * 16);
            aQ[mt][kh].q[1] = *(const uint4*)(qp + hi * 16 + 8);
        }

    v8f o[2][4];
    #pragma unroll
    for (int mt = 0; mt < 2; ++mt)
        #pragma unroll
        for (int nt = 0; nt < 4; ++nt)
            #pragma unroll
            for (int v = 0; v < 8; ++v) o[mt][nt][v] = 0.0f;

    float mi[2][8], li[2][8];
    #pragma unroll
    for (int mt = 0; mt < 2; ++mt)
        #pragma unroll
        for (int v = 0; v < 8; ++v) { mi[mt][v] = -1e30f; li[mt][v] = 0.0f; }

    for (int key0 = 0; key0 < N_SEQ; key0 += 32) {
        // ---- S = Qs . K^T for 32 keys (two 16-col tiles, two row-tiles) ----
        v8f s[2][2];
        #pragma unroll
        for (int mt = 0; mt < 2; ++mt)
            #pragma unroll
            for (int nt = 0; nt < 2; ++nt)
                #pragma unroll
                for (int v = 0; v < 8; ++v) s[mt][nt][v] = 0.0f;

        #pragma unroll
        for (int nt = 0; nt < 2; ++nt)
            #pragma unroll
            for (int kh = 0; kh < 2; ++kh) {
                const __bf16* kp =
                    Kh + (size_t)(key0 + nt * 16 + m) * DH + kh * 32 + hi * 16;
                Frag b;
                b.q[0] = *(const uint4*)(kp);
                b.q[1] = *(const uint4*)(kp + 8);
                #pragma unroll
                for (int mt = 0; mt < 2; ++mt)
                    s[mt][nt] = __builtin_amdgcn_wmma_f32_16x16x32_bf16(
                        false, aQ[mt][kh].v, false, b.v, (short)0,
                        s[mt][nt], false, false);
            }

        // ---- online softmax over this 32-key chunk ----
        #pragma unroll
        for (int mt = 0; mt < 2; ++mt)
            #pragma unroll
            for (int v = 0; v < 8; ++v) {
                float mx = fmaxf(s[mt][0][v], s[mt][1][v]);
                mx = fmaxf(mx, __shfl_xor(mx, 1, 32));
                mx = fmaxf(mx, __shfl_xor(mx, 2, 32));
                mx = fmaxf(mx, __shfl_xor(mx, 4, 32));
                mx = fmaxf(mx, __shfl_xor(mx, 8, 32));
                float mnew  = fmaxf(mi[mt][v], mx);
                float alpha = __expf(mi[mt][v] - mnew);
                float p0 = __expf(s[mt][0][v] - mnew);
                float p1 = __expf(s[mt][1][v] - mnew);
                float rs = p0 + p1;
                rs += __shfl_xor(rs, 1, 32);
                rs += __shfl_xor(rs, 2, 32);
                rs += __shfl_xor(rs, 4, 32);
                rs += __shfl_xor(rs, 8, 32);
                li[mt][v] = li[mt][v] * alpha + rs;
                mi[mt][v] = mnew;
                #pragma unroll
                for (int nt = 0; nt < 4; ++nt) o[mt][nt][v] *= alpha;
                int row = v + hi * 8;
                Pld[wave][mt][row * 32 + m]      = (__bf16)p0;
                Pld[wave][mt][row * 32 + m + 16] = (__bf16)p1;
            }
        // LDS is in-order within a wave; make the RAW explicit anyway.
        asm volatile("s_wait_dscnt 0x0" ::: "memory");

        // reload P as 16x32 A-fragments: two 16B chunks per lane per row-tile
        Frag pa[2];
        #pragma unroll
        for (int mt = 0; mt < 2; ++mt) {
            pa[mt].q[0] = *(const uint4*)&Pld[wave][mt][m * 32 + hi * 16];
            pa[mt].q[1] = *(const uint4*)&Pld[wave][mt][m * 32 + hi * 16 + 8];
        }

        // ---- O += P . V ---- (V col-major: key dim contiguous per lane)
        #pragma unroll
        for (int vt = 0; vt < 4; ++vt) {
            const __bf16* vp =
                Vth + (size_t)(vt * 16 + m) * N_SEQ + key0 + hi * 16;
            Frag b;
            b.q[0] = *(const uint4*)(vp);
            b.q[1] = *(const uint4*)(vp + 8);
            #pragma unroll
            for (int mt = 0; mt < 2; ++mt)
                o[mt][vt] = __builtin_amdgcn_wmma_f32_16x16x32_bf16(
                    false, pa[mt].v, false, b.v, (short)0, o[mt][vt], false, false);
        }
    }

    // normalize and store: out[n][h*64 + k]
    #pragma unroll
    for (int mt = 0; mt < 2; ++mt)
        #pragma unroll
        for (int vt = 0; vt < 4; ++vt)
            #pragma unroll
            for (int v = 0; v < 8; ++v) {
                int row = q0 + mt * 16 + v + hi * 8;
                Out[(size_t)row * DM + h * DH + vt * 16 + m] =
                    o[mt][vt][v] / li[mt][v];
            }
}

// ---------------------------------------------------------------------------
extern "C" void kernel_launch(void* const* d_in, const int* in_sizes, int n_in,
                              void* d_out, int out_size, void* d_ws, size_t ws_size,
                              hipStream_t stream) {
    const float* XQ = (const float*)d_in[0];
    const float* XK = (const float*)d_in[1];
    const float* XV = (const float*)d_in[2];
    const float* Wq = (const float*)d_in[3];
    const float* bq = (const float*)d_in[4];
    const float* Wk = (const float*)d_in[5];
    const float* bk = (const float*)d_in[6];
    const float* Wv = (const float*)d_in[7];
    const float* bv = (const float*)d_in[8];
    float* out = (float*)d_out;

    const size_t perQKV = (size_t)H_HEADS * N_SEQ * DH;   // 4M bf16 = 8 MB
    const size_t perX   = (size_t)N_SEQ * DM;             // 4M bf16 = 8 MB
    const size_t perW   = (size_t)H_HEADS * DH * DM;      // 1M bf16 = 2 MB

    __bf16* Qb   = (__bf16*)d_ws;
    __bf16* Kb   = Qb + perQKV;
    __bf16* Vt   = Kb + perQKV;
    __bf16* Xbuf = Vt + perQKV;      // reused serially for XQ, XK, XV
    __bf16* Wbuf = Xbuf + perX;      // reused serially for Wq, Wk, Wv
    // total: 3*8 + 8 + 2 = 34 MB

    dim3 blk(128);
    dim3 gridT(N_SEQ / 32 / 4, H_HEADS);       // 32 x 16 blocks, 32 rows/wave
    const int cvtxBlocks = (int)(perX / 8 / 256);    // 2048
    const int cvtwBlocks = (int)(perW / 8 / 256);    // 512
    const float inv_sqrt_dh = 0.125f;

    // Q (score scale folded into projection output)
    cvtx_kernel<<<cvtxBlocks, 256, 0, stream>>>(XQ, Xbuf);
    cvtw_kernel<<<cvtwBlocks, 256, 0, stream>>>(Wq, Wbuf);
    proj_kernel<<<gridT, blk, 0, stream>>>(Xbuf, Wbuf, bq, Qb, inv_sqrt_dh, 0);
    // K
    cvtx_kernel<<<cvtxBlocks, 256, 0, stream>>>(XK, Xbuf);
    cvtw_kernel<<<cvtwBlocks, 256, 0, stream>>>(Wk, Wbuf);
    proj_kernel<<<gridT, blk, 0, stream>>>(Xbuf, Wbuf, bk, Kb, 1.0f, 0);
    // V (stored transposed for contiguous B-fragments in attention)
    cvtx_kernel<<<cvtxBlocks, 256, 0, stream>>>(XV, Xbuf);
    cvtw_kernel<<<cvtwBlocks, 256, 0, stream>>>(Wv, Wbuf);
    proj_kernel<<<gridT, blk, 0, stream>>>(Xbuf, Wbuf, bv, Vt, 1.0f, 1);

    attn_kernel<<<gridT, blk, 0, stream>>>(Qb, Kb, Vt, out);
}
